// saCoronalBlock_53326313947745
// MI455X (gfx1250) — matrix operand order repaired and checked by reading, not compiled, not run on hardware
//
#include <hip/hip_runtime.h>

// CDNA5 / gfx1250 implementation of the coronal self-attention block.
// Key insight: the (0,3,2,1,4) permutation only reorders contraction axes,
// so S = X·Xtᵀ and out = att·G can be computed directly in the original
// [B,C,H,W,D] layout on contiguous 80x80 (w x d) slices.
// Math path: f32 -> f16 operands, V_WMMA_F32_16X16X32_F16, f32 accumulate.

typedef __attribute__((ext_vector_type(16))) _Float16 v16h;
typedef __attribute__((ext_vector_type(8)))  float    v8f;

#define TW 51          // LDS row stride in dwords (odd bank count, 48 data + pad)
#define B_STRIDE 16384000ull   // 32*80*80*80 floats per batch
#define GRAM_SLICES 10
#define APPLY_SLICES 8

union V16U { v16h v; unsigned int u[8]; };

__device__ inline unsigned short f2h_bits(float f) {
  _Float16 h = (_Float16)f;
  unsigned short u;
  __builtin_memcpy(&u, &h, 2);
  return u;
}
__device__ inline unsigned int pack2h(float a, float b) {
  return (unsigned int)f2h_bits(a) | ((unsigned int)f2h_bits(b) << 16);
}

// A operand: 16x32 f16, rows = row0..row0+15 of LDS tile, k = kbase..kbase+31.
// ISA layout: lane m = l%16; VGPR r holds k pair (r<4 ? 2r : 2r+8) + 8*(l/16).
__device__ inline v16h load_A(const unsigned int* tile, int row0, int kbase, int lane) {
  const int m  = lane & 15;
  const int kh = (lane >> 4) * 8;
  V16U r;
#pragma unroll
  for (int q = 0; q < 8; ++q) {
    const int klo = (q < 4 ? 2 * q : 2 * q + 8) + kh;
    r.u[q] = tile[(row0 + m) * TW + ((kbase + klo) >> 1)];
  }
  return r.v;
}

// B operand: 32x16 f16, cols = col0..col0+15 (tile rows), k along tile words.
// ISA layout: lane n = l%16; VGPR v holds k pair (v<4 ? 2v : 2v+8) + 8*(l/16).
__device__ inline v16h load_B(const unsigned int* tile, int col0, int kbase, int lane) {
  const int n  = lane & 15;
  const int kh = (lane >> 4) * 8;
  V16U r;
#pragma unroll
  for (int q = 0; q < 8; ++q) {
    const int klo = (q < 4 ? 2 * q : 2 * q + 8) + kh;
    r.u[q] = tile[(col0 + n) * TW + ((kbase + klo) >> 1)];
  }
  return r.v;
}

__global__ void __launch_bounds__(256) zero_kernel(float* __restrict__ s, int n) {
  int i = blockIdx.x * 256 + threadIdx.x;
  if (i < n) s[i] = 0.0f;
}

// ---------------- Gram: S[b] += slice · sliceᵀ over (c,h) slices ----------
__global__ void __launch_bounds__(256) gram_kernel(const float* __restrict__ x,
                                                   const float* __restrict__ xt,
                                                   float* __restrict__ S) {
  __shared__ unsigned int lx[80 * TW];
  __shared__ unsigned int lk[80 * TW];
  const int tid  = threadIdx.x;
  const int lane = tid & 31;
  const int wave = tid >> 5;
  const int b    = blockIdx.y;
  const size_t bOff = (size_t)b * B_STRIDE;

  // zero the k-padding words (d = 80..95 region + spare), done once
  for (int i = tid; i < 80 * 11; i += 256) {
    const int row = i / 11, w = 40 + i % 11;
    lx[row * TW + w] = 0u;
    lk[row * TW + w] = 0u;
  }

  const v8f vzero = {0.f, 0.f, 0.f, 0.f, 0.f, 0.f, 0.f, 0.f};
  v8f acc[4];
#pragma unroll
  for (int t = 0; t < 4; ++t) acc[t] = vzero;

  const int s0 = blockIdx.x * GRAM_SLICES;
  for (int si = 0; si < GRAM_SLICES; ++si) {
    const float* px = x  + bOff + (size_t)(s0 + si) * 6400;
    const float* pk = xt + bOff + (size_t)(s0 + si) * 6400;
    __syncthreads();  // previous compute done before overwriting tiles
    for (int idx = tid; idx < 3200; idx += 256) {
      const float2 vx = ((const float2*)px)[idx];
      const float2 vk = ((const float2*)pk)[idx];
      const int pos = (idx / 40) * TW + (idx % 40);
      lx[pos] = pack2h(vx.x, vx.y);
      lk[pos] = pack2h(vk.x, vk.y);
    }
    __syncthreads();
#pragma unroll
    for (int ti = 0; ti < 4; ++ti) {
      const int t = wave + ti * 8;
      if (t < 25) {
        const int i0 = (t / 5) * 16, j0 = (t % 5) * 16;
#pragma unroll
        for (int ks = 0; ks < 3; ++ks) {
          v16h a  = load_A(lx, i0, ks * 32, lane);
          v16h bb = load_B(lk, j0, ks * 32, lane);
          acc[ti] = __builtin_amdgcn_wmma_f32_16x16x32_f16(
              false, a, false, bb, (short)0, acc[ti], false, false);
        }
      }
    }
  }

  float* Sb = S + b * 6400;
  const int n  = lane & 15;
  const int mh = (lane >> 4) * 8;
#pragma unroll
  for (int ti = 0; ti < 4; ++ti) {
    const int t = wave + ti * 8;
    if (t < 25) {
      const int i0 = (t / 5) * 16, j0 = (t % 5) * 16;
#pragma unroll
      for (int r = 0; r < 8; ++r)
        atomicAdd(&Sb[(i0 + r + mh) * 80 + (j0 + n)], acc[ti][r]);
    }
  }
}

// ---------------- Row softmax, in place (tiny) ----------------------------
__global__ void __launch_bounds__(128) softmax_kernel(float* __restrict__ S) {
  __shared__ float red[128];
  float* p = S + blockIdx.x * 80;
  const int tid = threadIdx.x;
  float v = (tid < 80) ? p[tid] : -3.4e38f;
  red[tid] = v;
  __syncthreads();
  for (int s = 64; s; s >>= 1) {
    if (tid < s) red[tid] = fmaxf(red[tid], red[tid + s]);
    __syncthreads();
  }
  const float m = red[0];
  __syncthreads();
  const float e = (tid < 80) ? __expf(v - m) : 0.f;
  red[tid] = e;
  __syncthreads();
  for (int s = 64; s; s >>= 1) {
    if (tid < s) red[tid] += red[tid + s];
    __syncthreads();
  }
  const float inv = 1.f / red[0];
  if (tid < 80) p[tid] = e * inv;
}

// ---------------- Apply: out_slice = att[b] · g_slice ---------------------
__global__ void __launch_bounds__(256) apply_kernel(const float* __restrict__ att,
                                                    const float* __restrict__ g,
                                                    float* __restrict__ out) {
  __shared__ unsigned int la[80 * TW];  // att rows=w(i), words = j pairs
  __shared__ unsigned int lg[80 * TW];  // g transposed: rows=d, words = j pairs
  const int tid  = threadIdx.x;
  const int lane = tid & 31;
  const int wave = tid >> 5;
  const int b    = blockIdx.y;
  const size_t bOff = (size_t)b * B_STRIDE;

  for (int i = tid; i < 80 * 11; i += 256) {
    const int row = i / 11, w = 40 + i % 11;
    la[row * TW + w] = 0u;
    lg[row * TW + w] = 0u;
  }
  const float* A = att + b * 6400;
  for (int idx = tid; idx < 3200; idx += 256) {
    const float2 v = ((const float2*)A)[idx];
    la[(idx / 40) * TW + (idx % 40)] = pack2h(v.x, v.y);
  }

  unsigned short* lgh = (unsigned short*)lg;
  for (int si = 0; si < APPLY_SLICES; ++si) {
    const int s = blockIdx.x * APPLY_SLICES + si;
    const float* ps = g   + bOff + (size_t)s * 6400;
    float*       po = out + bOff + (size_t)s * 6400;
    __syncthreads();  // prior reads of lg done (also covers la init)
    for (int idx = tid; idx < 3200; idx += 256) {
      const int j = idx / 40, c = idx % 40;
      const float2 v = ((const float2*)ps)[idx];
      lgh[(2 * c) * (2 * TW) + j]     = f2h_bits(v.x);
      lgh[(2 * c + 1) * (2 * TW) + j] = f2h_bits(v.y);
    }
    __syncthreads();
    const int n  = lane & 15;
    const int mh = (lane >> 4) * 8;
    const v8f vzero = {0.f, 0.f, 0.f, 0.f, 0.f, 0.f, 0.f, 0.f};
#pragma unroll
    for (int ti = 0; ti < 4; ++ti) {
      const int t = wave + ti * 8;
      if (t < 25) {
        const int i0 = (t / 5) * 16, n0 = (t % 5) * 16;
        v8f acc = vzero;
#pragma unroll
        for (int ks = 0; ks < 3; ++ks) {
          v16h a  = load_A(la, i0, ks * 32, lane);
          v16h bb = load_B(lg, n0, ks * 32, lane);
          acc = __builtin_amdgcn_wmma_f32_16x16x32_f16(
              false, a, false, bb, (short)0, acc, false, false);
        }
#pragma unroll
        for (int r = 0; r < 8; ++r)
          po[(i0 + r + mh) * 80 + (n0 + n)] = acc[r];
      }
    }
  }
}

extern "C" void kernel_launch(void* const* d_in, const int* in_sizes, int n_in,
                              void* d_out, int out_size, void* d_ws, size_t ws_size,
                              hipStream_t stream) {
  const float* x  = (const float*)d_in[0];
  const float* xt = (const float*)d_in[1];
  const float* g  = (const float*)d_in[2];
  float* out = (float*)d_out;
  float* S   = (float*)d_ws;  // 2 * 80 * 80 floats = 51.2 KB

  zero_kernel<<<(2 * 6400 + 255) / 256, 256, 0, stream>>>(S, 2 * 6400);
  gram_kernel<<<dim3(2560 / GRAM_SLICES, 2), 256, 0, stream>>>(x, xt, S);
  softmax_kernel<<<2 * 80, 128, 0, stream>>>(S);
  apply_kernel<<<dim3(2560 / APPLY_SLICES, 2), 256, 0, stream>>>(S, g, out);
}
